// Deepsets_18657337934210
// MI455X (gfx1250) — compile-verified, matrix operand order checked by reading, not run on hardware
//
#include <hip/hip_runtime.h>
#include <hip/hip_bf16.h>

// ---------------------------------------------------------------------------
// DeepSets: h = relu(relu(x@W1+b1)@W2+b2); out[s] = mean over tokens in set s
// T=262144, D_IN=128, D_H=256, D_OUT=128, N_SETS=8192
// f16 WMMA (16x16x32) for both GEMMs. Each wave owns 32 tokens (2 M-tiles) so
// every B fragment load feeds 2 WMMAs; B fragments for a whole n-tile are
// batch-loaded into registers ahead of the WMMA chain so L2 latency overlaps
// matrix issue. Segment-sum via run-merged f32 atomics into d_out.
// ---------------------------------------------------------------------------

typedef __attribute__((ext_vector_type(16))) _Float16 v16h;
typedef __attribute__((ext_vector_type(8)))  _Float16 v8h;
typedef __attribute__((ext_vector_type(8)))  float    v8f;

#define T_TOKENS   262144
#define D_IN       128
#define D_H        256
#define D_OUT      128
#define N_SETS     8192

#define WAVES_PER_BLOCK 2
#define MTILES          2                                   // 16-row tiles per wave
#define TOK_PER_WAVE    (MTILES * 16)                       // 32
#define TOK_PER_BLOCK   (WAVES_PER_BLOCK * TOK_PER_WAVE)    // 64

#define KS1 (D_IN / 32)   // 4 k-steps, layer 1
#define KS2 (D_H / 32)    // 8 k-steps, layer 2
#define NT1 (D_H / 16)    // 16 n-tiles, layer 1
#define NT2 (D_OUT / 16)  // 8 n-tiles, layer 2

// LDS row stride in halves for the h tile (pad 256 -> 264: row-to-row bank
// rotation of 4, so 16 rows @ same k hit 16 distinct banks)
#define H_STRIDE 264

// ---------------------------------------------------------------------------
// 16-bit WMMA operand fragment layout (ISA 7.12.2), half index i in 0..15:
//   i in [0,8)  -> k_local = hi*8 + i          (K 0..7  / 8..15)
//   i in [8,16) -> k_local = 16 + hi*8 + (i-8) (K 16..23 / 24..31)
// A: lane16 = row M.  B (column-major consumption): lane16 = col N.
// ---------------------------------------------------------------------------
__device__ __forceinline__ int frag_klocal(int i, int hi) {
    return (i < 8) ? (hi * 8 + i) : (16 + hi * 8 + (i - 8));
}

// Pre-swizzle W1 [128x256] and W2 [256x128] (f32, [in,out]) into f16 B-operand
// fragments: ((nt*KS + ks)*32 + lane)*16 halves -> one contiguous 32B v16h per
// lane per WMMA.
__global__ void prep_weights_kernel(const float* __restrict__ W1,
                                    const float* __restrict__ W2,
                                    _Float16* __restrict__ w1f,
                                    _Float16* __restrict__ w2f) {
    int idx = blockIdx.x * blockDim.x + threadIdx.x;     // 0 .. 65535
    bool is2 = idx >= (D_IN * D_H);                      // second half -> W2
    int t = is2 ? idx - (D_IN * D_H) : idx;

    int i    = t & 15;
    int lane = (t >> 4) & 31;
    int fid  = t >> 9;                                   // nt*KS + ks
    int KS   = is2 ? KS2 : KS1;
    int ks   = fid % KS;
    int nt   = fid / KS;

    int lane16 = lane & 15;
    int hi     = lane >> 4;
    int k      = ks * 32 + frag_klocal(i, hi);
    int n      = nt * 16 + lane16;

    if (!is2) w1f[t] = (_Float16)W1[k * D_H + n];
    else      w2f[t] = (_Float16)W2[k * D_OUT + n];
}

__device__ __forceinline__ v8f wmma_f16(v16h a, v16h b, v8f c) {
    return __builtin_amdgcn_wmma_f32_16x16x32_f16(
        false, a, false, b, (short)0, c, false, false);
}

// Build one A fragment from two contiguous 8-float runs (f32 -> f16).
__device__ __forceinline__ v16h pack_a_from_f32(const float* p0, const float* p1) {
    const float4 f0 = *(const float4*)(p0);
    const float4 f1 = *(const float4*)(p0 + 4);
    const float4 f2 = *(const float4*)(p1);
    const float4 f3 = *(const float4*)(p1 + 4);
    v16h a;
    a[0]  = (_Float16)f0.x; a[1]  = (_Float16)f0.y;
    a[2]  = (_Float16)f0.z; a[3]  = (_Float16)f0.w;
    a[4]  = (_Float16)f1.x; a[5]  = (_Float16)f1.y;
    a[6]  = (_Float16)f1.z; a[7]  = (_Float16)f1.w;
    a[8]  = (_Float16)f2.x; a[9]  = (_Float16)f2.y;
    a[10] = (_Float16)f2.z; a[11] = (_Float16)f2.w;
    a[12] = (_Float16)f3.x; a[13] = (_Float16)f3.y;
    a[14] = (_Float16)f3.z; a[15] = (_Float16)f3.w;
    return a;
}

// Fused MLP + segment-sum accumulation. One wave owns 32 tokens (2 M-tiles).
__global__ void __launch_bounds__(WAVES_PER_BLOCK * 32)
deepsets_mlp_kernel(const float* __restrict__ x,
                    const int* __restrict__ seg,
                    const _Float16* __restrict__ w1f,
                    const _Float16* __restrict__ w2f,
                    const float* __restrict__ b1,
                    const float* __restrict__ b2,
                    float* __restrict__ outSums) {
    __shared__ _Float16 sH[WAVES_PER_BLOCK][TOK_PER_WAVE * H_STRIDE]; // 33.8 KB

    const int wave   = threadIdx.x >> 5;
    const int lane   = threadIdx.x & 31;
    const int lane16 = lane & 15;
    const int hi     = lane >> 4;

    const int tokBase = blockIdx.x * TOK_PER_BLOCK + wave * TOK_PER_WAVE;

    // ------------------------- Layer 1: A from global x -------------------
    v16h a1[MTILES][KS1];
#pragma unroll
    for (int mt = 0; mt < MTILES; ++mt) {
        const float* xrow = x + (size_t)(tokBase + mt * 16 + lane16) * D_IN;
        if (mt == 0)   // hint the next block's tile (gfx1250 global_prefetch_b8)
            __builtin_prefetch(xrow + (size_t)TOK_PER_BLOCK * D_IN, 0, 1);
#pragma unroll
        for (int ks = 0; ks < KS1; ++ks) {
            const int k0 = ks * 32 + hi * 8;
            a1[mt][ks] = pack_a_from_f32(xrow + k0, xrow + k0 + 16);
        }
    }

    // 16 n-tiles of h; batch all 4 B fragments first, then 8 WMMAs.
#pragma unroll
    for (int nt = 0; nt < NT1; ++nt) {
        v16h vb[KS1];
#pragma unroll
        for (int ks = 0; ks < KS1; ++ks)
            vb[ks] = *(const v16h*)(w1f + ((size_t)(nt * KS1 + ks) * 32 + lane) * 16);

        v8f c[MTILES];
#pragma unroll
        for (int mt = 0; mt < MTILES; ++mt) c[mt] = (v8f){};
#pragma unroll
        for (int ks = 0; ks < KS1; ++ks)
#pragma unroll
            for (int mt = 0; mt < MTILES; ++mt)
                c[mt] = wmma_f16(a1[mt][ks], vb[ks], c[mt]);

        const int n = nt * 16 + lane16;
        const float bias = b1[n];
#pragma unroll
        for (int mt = 0; mt < MTILES; ++mt) {
#pragma unroll
            for (int r = 0; r < 8; ++r) {
                float v = c[mt][r] + bias;
                v = v > 0.0f ? v : 0.0f;
                sH[wave][(mt * 16 + r + 8 * hi) * H_STRIDE + n] = (_Float16)v;
            }
        }
    }

    __syncthreads();   // order LDS stores before fragment reads

    // ------------------------- Layer 2: A from LDS h ----------------------
    v16h a2[MTILES][KS2];
#pragma unroll
    for (int mt = 0; mt < MTILES; ++mt) {
#pragma unroll
        for (int ks = 0; ks < KS2; ++ks) {
            const _Float16* base =
                &sH[wave][(mt * 16 + lane16) * H_STRIDE + ks * 32 + hi * 8];
            const v8h lo = *(const v8h*)(base);
            const v8h up = *(const v8h*)(base + 16);
            v16h a;
#pragma unroll
            for (int i = 0; i < 8; ++i) { a[i] = lo[i]; a[i + 8] = up[i]; }
            a2[mt][ks] = a;
        }
    }

    // Segment ids for the 8 token rows per M-tile this lane emits.
    int sid[MTILES][8];
#pragma unroll
    for (int mt = 0; mt < MTILES; ++mt)
#pragma unroll
        for (int r = 0; r < 8; ++r)
            sid[mt][r] = seg[tokBase + mt * 16 + r + 8 * hi];

#pragma unroll
    for (int nt = 0; nt < NT2; ++nt) {
        v16h vb[KS2];
#pragma unroll
        for (int ks = 0; ks < KS2; ++ks)
            vb[ks] = *(const v16h*)(w2f + ((size_t)(nt * KS2 + ks) * 32 + lane) * 16);

        v8f c[MTILES];
#pragma unroll
        for (int mt = 0; mt < MTILES; ++mt) c[mt] = (v8f){};
#pragma unroll
        for (int ks = 0; ks < KS2; ++ks)
#pragma unroll
            for (int mt = 0; mt < MTILES; ++mt)
                c[mt] = wmma_f16(a2[mt][ks], vb[ks], c[mt]);

        const int n = nt * 16 + lane16;
        const float bias = b2[n];
        // Run-merge atomics: segment_ids are sorted, so consecutive rows
        // (consecutive tokens) usually share a segment.
#pragma unroll
        for (int mt = 0; mt < MTILES; ++mt) {
            float acc = 0.0f;
            int   cur = sid[mt][0];
#pragma unroll
            for (int r = 0; r < 8; ++r) {
                float v = c[mt][r] + bias;
                v = v > 0.0f ? v : 0.0f;
                if (sid[mt][r] != cur) {
                    atomicAdd(outSums + (size_t)cur * D_OUT + n, acc);
                    acc = 0.0f;
                    cur = sid[mt][r];
                }
                acc += v;
            }
            atomicAdd(outSums + (size_t)cur * D_OUT + n, acc);
        }
    }
}

__global__ void count_kernel(const int* __restrict__ seg, float* __restrict__ counts) {
    int t = blockIdx.x * blockDim.x + threadIdx.x;
    if (t < T_TOKENS) atomicAdd(&counts[seg[t]], 1.0f);
}

__global__ void finalize_kernel(float* __restrict__ out, const float* __restrict__ counts) {
    int i = blockIdx.x * blockDim.x + threadIdx.x;
    if (i < N_SETS * D_OUT) {
        out[i] = out[i] / fmaxf(counts[i >> 7], 1.0f);
    }
}

extern "C" void kernel_launch(void* const* d_in, const int* in_sizes, int n_in,
                              void* d_out, int out_size, void* d_ws, size_t ws_size,
                              hipStream_t stream) {
    (void)in_sizes; (void)n_in; (void)ws_size;

    const float* x   = (const float*)d_in[0];
    const int*   seg = (const int*)d_in[1];
    // d_in[2] = num_segments scalar (known: 8192)
    const float* W1  = (const float*)d_in[3];
    const float* b1  = (const float*)d_in[4];
    const float* W2  = (const float*)d_in[5];
    const float* b2  = (const float*)d_in[6];

    float* out = (float*)d_out;

    // Workspace: [0,64KB) w1 frags, [64KB,128KB) w2 frags, [128KB,+32KB) counts
    _Float16* w1f    = (_Float16*)d_ws;
    _Float16* w2f    = (_Float16*)((char*)d_ws + (size_t)D_IN * D_H * 2);
    float*    counts = (float*)((char*)d_ws + (size_t)(D_IN * D_H + D_H * D_OUT) * 2);

    // Zero sums (in d_out) and counts every launch (graph-capture safe).
    hipMemsetAsync(out, 0, (size_t)out_size * sizeof(float), stream);
    hipMemsetAsync(counts, 0, (size_t)N_SETS * sizeof(float), stream);

    // Swizzle weights into WMMA B-fragment order (65536 elements total).
    prep_weights_kernel<<<(D_IN * D_H + D_H * D_OUT) / 256, 256, 0, stream>>>(W1, W2, w1f, w2f);

    // Fused MLP + segment-sum.
    deepsets_mlp_kernel<<<T_TOKENS / TOK_PER_BLOCK, WAVES_PER_BLOCK * 32, 0, stream>>>(
        x, seg, w1f, w2f, b1, b2, out);

    // Token counts per segment.
    count_kernel<<<T_TOKENS / 256, 256, 0, stream>>>(seg, counts);

    // Divide sums by counts in place.
    finalize_kernel<<<(N_SETS * D_OUT) / 256, 256, 0, stream>>>(out, counts);
}
